// ReactivityScoring_6820408066819
// MI455X (gfx1250) — matrix operand order repaired and checked by reading, not compiled, not run on hardware
//
#include <hip/hip_runtime.h>

typedef _Float16 h16;
typedef __attribute__((ext_vector_type(16))) _Float16 v16h;
typedef __attribute__((ext_vector_type(8)))  float    v8f;

#define H_DIM 128
#define PTOT  640000
#define BINF_DIM 11
#define NWAVES 8
#define TILES_PER_WAVE 8
#define ROWS_PER_BLOCK (NWAVES * TILES_PER_WAVE * 16) /* 1024 */
#define NBLOCKS (PTOT / ROWS_PER_BLOCK)               /* 625  */

// K-steps: 0..3 = W_local (K=0..127), 4..7 = W_global, 8 = [W_binary | b_global | 0]
#define NKSTEPS 9

__global__ __launch_bounds__(256)
void reactivity_kernel(const float* __restrict__ local_pair,
                       const float* __restrict__ global_pair,
                       const float* __restrict__ binary,
                       const int*   __restrict__ sidx,
                       const float* __restrict__ W_local,
                       const float* __restrict__ W_global,
                       const float* __restrict__ b_global,
                       const float* __restrict__ W_binary,
                       const float* __restrict__ W_score,
                       const float* __restrict__ b_score,
                       float* __restrict__ out)
{
    // Weights pre-swizzled to CDNA5 WMMA B-fragment layout: [kstep][ntile][lane][q]
    __shared__ h16   s_wfrag[NKSTEPS][8][32][16]; // 72 KB
    __shared__ float s_wscore[H_DIM * 5];         // 2.5 KB
    __shared__ float s_bscore[8];
    __shared__ float s_bfeat[NWAVES][16][12];     // 6 KB (per-wave gather staging)

    const int tid = threadIdx.x;

    // ---- stage weights (once per block) ----
    // B-fragment layout (16-bit B, 32x16 per k-step): lane = N within tile,
// halves q<8 hold K = 8*(lane/16) + q, q>=8 hold K = 16 + 8*(lane/16) + (q-8) within the step.
    for (int i = tid; i < NKSTEPS * 8 * 32 * 16; i += 256) {
        const int q  = i & 15;
        const int ln = (i >> 4) & 31;
        const int j  = (i >> 9) & 7;
        const int ks = i >> 12;                 // 0..8
        const int kk = 8 * (ln >> 4) + (q < 8 ? q : q + 8); // K within step, 0..31
        const int n  = 16 * j + (ln & 15);
        float v;
        if (ks < 4)       v = W_local [(32 * ks + kk) * H_DIM + n];
        else if (ks < 8)  v = W_global[(32 * (ks - 4) + kk) * H_DIM + n];
        else              v = (kk < BINF_DIM) ? W_binary[kk * H_DIM + n]
                             : (kk == BINF_DIM ? b_global[n] : 0.0f);
        s_wfrag[ks][j][ln][q] = (h16)v;
    }
    for (int i = tid; i < H_DIM * 5; i += 256) s_wscore[i] = W_score[i];
    if (tid < 5) s_bscore[tid] = b_score[tid];
    __syncthreads();

    const int wave = tid >> 5;
    const int lane = tid & 31;
    const int u    = lane & 15;   // column-within-tile / A-row id
    const int hi   = lane >> 4;   // lane half (K-interleave select)

    #pragma unroll 1
    for (int tile = 0; tile < TILES_PER_WAVE; ++tile) {
        const int p_base = blockIdx.x * ROWS_PER_BLOCK
                         + wave * (TILES_PER_WAVE * 16) + tile * 16;

        // ---- gather binary feats for this wave's 16 rows (lanes 0-15) ----
        if (hi == 0) {
            const int row = p_base + u;
            const int b  = sidx[3 * row + 0];
            const int i2 = sidx[3 * row + 1];
            const int j2 = sidx[3 * row + 2];
            const float* bp = binary + (size_t)((b * 100 + i2) * 100 + j2) * BINF_DIM;
            #pragma unroll
            for (int q = 0; q < BINF_DIM; ++q) s_bfeat[wave][u][q] = bp[q];
        }

        // ---- A fragments (f32 -> f16), CDNA5 16-bit A 16x32 layout ----
        // halves q<8 -> K = 32s + 8*hi + q ; q>=8 -> K = 32s + 8*hi + 8 + q
        // => two contiguous 8-float runs per k-step: [base..base+7], [base+16..base+23]
        const float* lrow = local_pair  + (size_t)(p_base + u) * H_DIM;
        const float* grow = global_pair + (size_t)(p_base + u) * H_DIM;
        v16h aL[4], aG[4];
        #pragma unroll
        for (int s = 0; s < 4; ++s) {
            const int base = 32 * s + 8 * hi;
            const float4 l0 = *(const float4*)(lrow + base);
            const float4 l1 = *(const float4*)(lrow + base + 4);
            const float4 l2 = *(const float4*)(lrow + base + 16);
            const float4 l3 = *(const float4*)(lrow + base + 20);
            const float4 g0 = *(const float4*)(grow + base);
            const float4 g1 = *(const float4*)(grow + base + 4);
            const float4 g2 = *(const float4*)(grow + base + 16);
            const float4 g3 = *(const float4*)(grow + base + 20);
            aL[s][0]=(h16)l0.x; aL[s][1]=(h16)l0.y; aL[s][2]=(h16)l0.z; aL[s][3]=(h16)l0.w;
            aL[s][4]=(h16)l1.x; aL[s][5]=(h16)l1.y; aL[s][6]=(h16)l1.z; aL[s][7]=(h16)l1.w;
            aL[s][8]=(h16)l2.x; aL[s][9]=(h16)l2.y; aL[s][10]=(h16)l2.z; aL[s][11]=(h16)l2.w;
            aL[s][12]=(h16)l3.x; aL[s][13]=(h16)l3.y; aL[s][14]=(h16)l3.z; aL[s][15]=(h16)l3.w;
            aG[s][0]=(h16)g0.x; aG[s][1]=(h16)g0.y; aG[s][2]=(h16)g0.z; aG[s][3]=(h16)g0.w;
            aG[s][4]=(h16)g1.x; aG[s][5]=(h16)g1.y; aG[s][6]=(h16)g1.z; aG[s][7]=(h16)g1.w;
            aG[s][8]=(h16)g2.x; aG[s][9]=(h16)g2.y; aG[s][10]=(h16)g2.z; aG[s][11]=(h16)g2.w;
            aG[s][12]=(h16)g3.x; aG[s][13]=(h16)g3.y; aG[s][14]=(h16)g3.z; aG[s][15]=(h16)g3.w;
        }

        // ---- binary+bias A fragment: [bf(0..10) | 1.0 | zeros], same A layout ----
        v16h aB;
        #pragma unroll
        for (int q = 0; q < 16; ++q) {
            const int K = 8 * hi + (q < 8 ? q : q + 8);
            float v;
            if (K < BINF_DIM)        v = s_bfeat[wave][u][K];
            else if (K == BINF_DIM)  v = 1.0f;    // picks up b_global row
            else                     v = 0.0f;
            aB[q] = (h16)v;
        }

        // prefetch next tile's A rows (global_prefetch_b8)
        if (tile + 1 < TILES_PER_WAVE) {
            __builtin_prefetch(lrow + 16 * H_DIM, 0, 1);
            __builtin_prefetch(grow + 16 * H_DIM, 0, 1);
        }

        float partial[8][5];
        #pragma unroll
        for (int v = 0; v < 8; ++v)
            #pragma unroll
            for (int s5 = 0; s5 < 5; ++s5) partial[v][s5] = 0.f;

        // ---- 8 N-tiles x 9 K-steps; keep the loop rolled so only one tile's
        // B fragments are live at a time (prevents scratch spilling).
        #pragma unroll 1
        for (int j = 0; j < 8; ++j) {
            const int n = 16 * j + u;
            // two independent accumulation chains to halve dependent-WMMA depth
            v8f acc0 = {};
            v8f acc1 = {};
            #pragma unroll
            for (int s = 0; s < 4; ++s) {
                const v16h bL = *(const v16h*)&s_wfrag[s][j][lane][0];
                acc0 = __builtin_amdgcn_wmma_f32_16x16x32_f16(false, aL[s], false, bL,
                                                              (short)0, acc0, false, false);
                const v16h bG = *(const v16h*)&s_wfrag[4 + s][j][lane][0];
                acc1 = __builtin_amdgcn_wmma_f32_16x16x32_f16(false, aG[s], false, bG,
                                                              (short)0, acc1, false, false);
            }
            const v16h bB = *(const v16h*)&s_wfrag[8][j][lane][0];
            acc0 = __builtin_amdgcn_wmma_f32_16x16x32_f16(false, aB, false, bB,
                                                          (short)0, acc0, false, false);

            // epilogue: merge chains, ReLU, score partials
            #pragma unroll
            for (int v = 0; v < 8; ++v) {
                const float f = fmaxf(acc0[v] + acc1[v], 0.f);
                #pragma unroll
                for (int s5 = 0; s5 < 5; ++s5) partial[v][s5] += f * s_wscore[n * 5 + s5];
            }
        }

        // ---- reduce score partials across the 16 lanes sharing a row set ----
        #pragma unroll
        for (int mask = 1; mask <= 8; mask <<= 1) {
            #pragma unroll
            for (int v = 0; v < 8; ++v)
                #pragma unroll
                for (int s5 = 0; s5 < 5; ++s5)
                    partial[v][s5] += __shfl_xor(partial[v][s5], mask, 32);
        }

        if (u < 8) {
            const size_t row = (size_t)p_base + 8 * hi + u;
            #pragma unroll
            for (int s5 = 0; s5 < 5; ++s5)
                out[row * 5 + s5] = partial[u][s5] + s_bscore[s5];
        }
    }
}

extern "C" void kernel_launch(void* const* d_in, const int* in_sizes, int n_in,
                              void* d_out, int out_size, void* d_ws, size_t ws_size,
                              hipStream_t stream) {
    const float* local_pair  = (const float*)d_in[0];
    const float* global_pair = (const float*)d_in[1];
    const float* binary      = (const float*)d_in[2];
    const int*   sidx        = (const int*)d_in[3];
    const float* W_local     = (const float*)d_in[4];
    const float* W_global    = (const float*)d_in[5];
    const float* b_global    = (const float*)d_in[6];
    const float* W_binary    = (const float*)d_in[7];
    const float* W_score     = (const float*)d_in[8];
    const float* b_score     = (const float*)d_in[9];
    float* out = (float*)d_out;

    reactivity_kernel<<<NBLOCKS, 256, 0, stream>>>(
        local_pair, global_pair, binary, sidx,
        W_local, W_global, b_global, W_binary, W_score, b_score, out);
}
